// Reinsertion_7739531067800
// MI455X (gfx1250) — matrix operand order repaired and checked by reading, not compiled, not run on hardware
//
#include <hip/hip_runtime.h>

#define B_  32
#define G_  512
#define D_  128
#define H_  4
#define KD_ 32
#define NORM_ 0.17677669529663687f  // 1/sqrt(32)

typedef __attribute__((ext_vector_type(16))) _Float16 v16h;
typedef __attribute__((ext_vector_type(8)))  float    v8f;

// ---------------------------------------------------------------------------
// Kernel 1: per-batch query projections. q[b, vec, h, k], NORM folded in.
// vec: 0 = pickup@Wq1, 1 = pickup@Wq2, 2 = delivery@Wq1, 3 = delivery@Wq2
// ---------------------------------------------------------------------------
__global__ void qproj_kernel(const float* __restrict__ h,
                             const int* __restrict__ pos_pickup,
                             const int* __restrict__ pos_delivery,
                             const float* __restrict__ Wq1,
                             const float* __restrict__ Wq2,
                             float* __restrict__ qbuf) {
  int b = blockIdx.x;
  int t = threadIdx.x;                 // 0..127 == m*32 + k
  int m = t >> 5, k = t & 31;
  const float* hp = h + ((size_t)b * G_ + pos_pickup[b]) * D_;
  const float* hd = h + ((size_t)b * G_ + pos_delivery[b]) * D_;
  float p1 = 0.f, p2 = 0.f, d1 = 0.f, d2 = 0.f;
  for (int d = 0; d < D_; ++d) {
    float w1 = Wq1[((size_t)m * D_ + d) * KD_ + k];
    float w2 = Wq2[((size_t)m * D_ + d) * KD_ + k];
    float hpv = hp[d], hdv = hd[d];
    p1 += hpv * w1; p2 += hpv * w2;
    d1 += hdv * w1; d2 += hdv * w2;
  }
  float* q = qbuf + b * 512;
  q[0 * 128 + t] = p1 * NORM_;
  q[1 * 128 + t] = p2 * NORM_;
  q[2 * 128 + t] = d1 * NORM_;
  q[3 * 128 + t] = d2 * NORM_;
}

// ---------------------------------------------------------------------------
// Kernel 2: per-node (b,g) compat scores -> folded fc1 half-rows A,C (32 each)
// One wave32 per node; lane l owns K-dim index k=l of each head.
// ---------------------------------------------------------------------------
__global__ void node_kernel(const float* __restrict__ h,
                            const int* __restrict__ rec,
                            const float* __restrict__ Wk1,
                            const float* __restrict__ Wk2,
                            const float* __restrict__ fc1_w,
                            const float* __restrict__ fc1_b,
                            const float* __restrict__ qbuf,
                            float* __restrict__ Abuf,
                            float* __restrict__ Cbuf) {
  int wave = threadIdx.x >> 5;
  int lane = threadIdx.x & 31;
  int node = blockIdx.x * 8 + wave;    // 0..16383
  int b = node >> 9;
  int gn = rec[node];
  const float* hr = h + (size_t)node * D_;
  const float* nr = h + ((size_t)b * G_ + gn) * D_;

  float K1[4] = {0.f, 0.f, 0.f, 0.f};
  float K2[4] = {0.f, 0.f, 0.f, 0.f};
  for (int d = 0; d < D_; ++d) {
    float hv = hr[d], nv = nr[d];
#pragma unroll
    for (int m = 0; m < 4; ++m) {
      K1[m] += hv * Wk1[((size_t)m * D_ + d) * KD_ + lane];
      K2[m] += nv * Wk2[((size_t)m * D_ + d) * KD_ + lane];
    }
  }

  const float* q = qbuf + b * 512;
  float c16[16];
#pragma unroll
  for (int m = 0; m < 4; ++m) {
    c16[0 + m]  = q[0 * 128 + m * 32 + lane] * K1[m];  // c_pp
    c16[4 + m]  = q[1 * 128 + m * 32 + lane] * K2[m];  // c_po
    c16[8 + m]  = q[2 * 128 + m * 32 + lane] * K1[m];  // c_dp
    c16[12 + m] = q[3 * 128 + m * 32 + lane] * K2[m];  // c_do
  }
#pragma unroll
  for (int mask = 1; mask < 32; mask <<= 1) {
#pragma unroll
    for (int s = 0; s < 16; ++s)
      c16[s] += __shfl_xor(c16[s], mask, 32);
  }

  // fold through fc1: A uses rows 0..7 (pp,po); C uses rows 8..15 (dp,do) + b1
  float a = 0.f, c = fc1_b[lane];
#pragma unroll
  for (int s = 0; s < 8; ++s) {
    a += c16[s]     * fc1_w[s * 32 + lane];
    c += c16[8 + s] * fc1_w[(8 + s) * 32 + lane];
  }
  Abuf[(size_t)node * 32 + lane] = a;
  Cbuf[(size_t)node * 32 + lane] = c;
}

// ---------------------------------------------------------------------------
// Kernel 3: the big (B,G,G) MLP via WMMA, transposed GEMM:
//   Z2^T = (fc2_w^T) x Z1^T  so the fc3 reduction is in-register.
// A-matrix  = fc2 weights (M=16 out-feats, K=32 in-feats), loop-invariant.
// B-matrix  = activations (K=32 feats, N=16 j-samples); lane holds sample
//             lm = lane&15, contiguous features half*16..half*16+15.
// D         = lane: (sample lm) x (out-feats r+8*half) in 8 VGPRs.
// One wave handles (b, j-tile of 16, i-chunk of 128), i unrolled by 2.
// ---------------------------------------------------------------------------
__global__ void __launch_bounds__(256) mlp_kernel(
    const float* __restrict__ Abuf, const float* __restrict__ Cbuf,
    const float* __restrict__ fc2_w, const float* __restrict__ fc2_b,
    const float* __restrict__ fc3_w, const float* __restrict__ fc3_b,
    float* __restrict__ out) {
  int wave = threadIdx.x >> 5;
  int lane = threadIdx.x & 31;
  int task = blockIdx.x * 8 + wave;    // 0..4095 = b(32) x jt(32) x ic(4)
  int b  = task >> 7;
  int jt = (task >> 2) & 31;
  int ic = task & 3;
  int jbase = jt * 16;
  int ibase = ic * 128;
  int half = lane >> 4;                // which feature-half this lane holds
  int lm   = lane & 15;

  // fc2 weights^T as the f16 A operand, in this code's contiguous k order (same as B):
  // row M = lm (output feature), lanes<16 hold K {0..15},
  // lanes>=16 hold K {16..31}. wa0 -> out-feats 0..15, wa1 -> 16..31.
  v16h wa0, wa1;
#pragma unroll
  for (int t = 0; t < 16; ++t) {
    int k = half * 16 + t;
    wa0[t] = (_Float16)fc2_w[k * 32 + lm];
    wa1[t] = (_Float16)fc2_w[k * 32 + 16 + lm];
  }
  // fc2 bias into C accumulator: rows are output features r + 8*half
  v8f c0, c1;
#pragma unroll
  for (int r = 0; r < 8; ++r) {
    c0[r] = fc2_b[r + half * 8];
    c1[r] = fc2_b[16 + r + half * 8];
  }
  // fc3 weights for the output features this lane's D rows cover
  float w3a[8], w3b[8];
#pragma unroll
  for (int r = 0; r < 8; ++r) {
    w3a[r] = fc3_w[r + half * 8];
    w3b[r] = fc3_w[16 + r + half * 8];
  }
  float b3 = fc3_b[0];

  // C activations: sample j = jbase + lm, contiguous features half*16..+15
  const float4* cp =
      (const float4*)(Cbuf + ((size_t)b * G_ + jbase + lm) * 32 + half * 16);
  float cr[16];
  {
    float4 x0 = cp[0], x1 = cp[1], x2 = cp[2], x3 = cp[3];
    cr[0]=x0.x;  cr[1]=x0.y;  cr[2]=x0.z;  cr[3]=x0.w;
    cr[4]=x1.x;  cr[5]=x1.y;  cr[6]=x1.z;  cr[7]=x1.w;
    cr[8]=x2.x;  cr[9]=x2.y;  cr[10]=x2.z; cr[11]=x2.w;
    cr[12]=x3.x; cr[13]=x3.y; cr[14]=x3.z; cr[15]=x3.w;
  }

  for (int i = ibase; i < ibase + 128; i += 2) {
    const float* Arow0 = Abuf + ((size_t)b * G_ + i) * 32 + half * 16;
    const float* Arow1 = Arow0 + 32;
    __builtin_prefetch(Arow1 + 32, 0, 1);  // next pair's A rows
    const float4* ap0 = (const float4*)Arow0;
    const float4* ap1 = (const float4*)Arow1;
    float4 a0 = ap0[0], a1 = ap0[1], a2 = ap0[2], a3 = ap0[3];
    float4 e0 = ap1[0], e1 = ap1[1], e2 = ap1[2], e3 = ap1[3];
    float ar0[16] = {a0.x,a0.y,a0.z,a0.w, a1.x,a1.y,a1.z,a1.w,
                     a2.x,a2.y,a2.z,a2.w, a3.x,a3.y,a3.z,a3.w};
    float ar1[16] = {e0.x,e0.y,e0.z,e0.w, e1.x,e1.y,e1.z,e1.w,
                     e2.x,e2.y,e2.z,e2.w, e3.x,e3.y,e3.z,e3.w};

    // z1 = relu(A[i] + C[j]) packed into B-matrix f16 layout (contiguous K)
    v16h bv0, bv1;
#pragma unroll
    for (int t = 0; t < 16; ++t) {
      bv0[t] = (_Float16)fmaxf(ar0[t] + cr[t], 0.f);
      bv1[t] = (_Float16)fmaxf(ar1[t] + cr[t], 0.f);
    }

    // fc2 (transposed): 4 WMMAs for the two i rows x two out-feature halves
    v8f d00 = __builtin_amdgcn_wmma_f32_16x16x32_f16(
        false, wa0, false, bv0, (short)0, c0, false, false);
    v8f d01 = __builtin_amdgcn_wmma_f32_16x16x32_f16(
        false, wa1, false, bv0, (short)0, c1, false, false);
    v8f d10 = __builtin_amdgcn_wmma_f32_16x16x32_f16(
        false, wa0, false, bv1, (short)0, c0, false, false);
    v8f d11 = __builtin_amdgcn_wmma_f32_16x16x32_f16(
        false, wa1, false, bv1, (short)0, c1, false, false);

    // fc3: relu + weighted sum over this lane's 16 out-features (in-register)
    float p0 = 0.f, p1 = 0.f;
#pragma unroll
    for (int r = 0; r < 8; ++r) {
      p0 += fmaxf(d00[r], 0.f) * w3a[r] + fmaxf(d01[r], 0.f) * w3b[r];
      p1 += fmaxf(d10[r], 0.f) * w3a[r] + fmaxf(d11[r], 0.f) * w3b[r];
    }
    // combine the two feature halves: lanes L and L^16 hold complements
    p0 += __shfl_xor(p0, 16, 32);
    p1 += __shfl_xor(p1, 16, 32);

    // lanes 0-15 store row i, lanes 16-31 store row i+1 (all lanes active)
    float res = (half ? p1 : p0) + b3;
    out[((size_t)b * G_ + i + half) * G_ + jbase + lm] = res;
  }
}

// ---------------------------------------------------------------------------
extern "C" void kernel_launch(void* const* d_in, const int* in_sizes, int n_in,
                              void* d_out, int out_size, void* d_ws, size_t ws_size,
                              hipStream_t stream) {
  const float* h     = (const float*)d_in[0];
  const int*   pp    = (const int*)d_in[1];
  const int*   pd    = (const int*)d_in[2];
  const int*   rec   = (const int*)d_in[3];
  const float* Wq1   = (const float*)d_in[4];
  const float* Wk1   = (const float*)d_in[5];
  const float* Wq2   = (const float*)d_in[6];
  const float* Wk2   = (const float*)d_in[7];
  const float* fc1_w = (const float*)d_in[8];
  const float* fc1_b = (const float*)d_in[9];
  const float* fc2_w = (const float*)d_in[10];
  const float* fc2_b = (const float*)d_in[11];
  const float* fc3_w = (const float*)d_in[12];
  const float* fc3_b = (const float*)d_in[13];
  float* out = (float*)d_out;

  float* ws   = (float*)d_ws;
  float* qbuf = ws;                       // 32*512            = 16384 floats
  float* Abuf = qbuf + 32 * 512;          // 32*512*32         = 524288 floats
  float* Cbuf = Abuf + 32 * 512 * 32;     // 32*512*32         = 524288 floats

  qproj_kernel<<<dim3(B_), dim3(128), 0, stream>>>(h, pp, pd, Wq1, Wq2, qbuf);
  node_kernel<<<dim3((B_ * G_) / 8), dim3(256), 0, stream>>>(
      h, rec, Wk1, Wk2, fc1_w, fc1_b, qbuf, Abuf, Cbuf);
  mlp_kernel<<<dim3(512), dim3(256), 0, stream>>>(
      Abuf, Cbuf, fc2_w, fc2_b, fc3_w, fc3_b, out);
}
